// GravConv_49761491092127
// MI455X (gfx1250) — compile-verified, matrix operand order checked by reading, not run on hardware
//
#include <hip/hip_runtime.h>
#include <hip/hip_bf16.h>
#include <math.h>

#define N_NODES 40000
#define H_DIM   128
#define EMB     8
#define E_EDGES 960000
#define H1      129      // H+1
#define H1P     132      // padded stride for f32 h / accumulators
#define K0PAD   160      // 129 -> 160 (5 * 32)
#define KYPAD   544      // 516 -> 544 (17 * 32)
#define LN_EPS  1e-5f

typedef __attribute__((ext_vector_type(16))) __bf16 v16bf;
typedef __attribute__((ext_vector_type(8)))  __bf16 v8bf;
typedef __attribute__((ext_vector_type(8)))  float  v8f;

static __device__ __forceinline__ unsigned short f2bf(float f) {
    unsigned u = __float_as_uint(f);
    u += 0x7FFFu + ((u >> 16) & 1u);          // round-to-nearest-even
    return (unsigned short)(u >> 16);
}

// ---------------- zero init (accumulators; 0 is also the encoded -max) ------
__global__ void zero_f32(float* __restrict__ p, long long n) {
    long long i = (long long)blockIdx.x * blockDim.x + threadIdx.x;
    long long stride = (long long)gridDim.x * blockDim.x;
    for (; i < n; i += stride) p[i] = 0.0f;
}

// ---------------- build h = [hf, mean(hf)] in f32 (stride 132) + bf16 (160) -
__global__ __launch_bounds__(128)
void prep_h(const float* __restrict__ hf, float* __restrict__ h32,
            unsigned short* __restrict__ hbf) {
    int n = blockIdx.x, t = threadIdx.x;
    __shared__ float red[128];
    float v = hf[(size_t)n * 128 + t];
    red[t] = v;
    __syncthreads();
    for (int s = 64; s > 0; s >>= 1) {
        if (t < s) red[t] += red[t + s];
        __syncthreads();
    }
    float mean = red[0] * (1.0f / 128.0f);
    h32[(size_t)n * H1P + t] = v;
    hbf[(size_t)n * K0PAD + t] = f2bf(v);
    if (t == 0) {
        h32[(size_t)n * H1P + 128] = mean;
        h32[(size_t)n * H1P + 129] = 0.0f;
        h32[(size_t)n * H1P + 130] = 0.0f;
        h32[(size_t)n * H1P + 131] = 0.0f;
        hbf[(size_t)n * K0PAD + 128] = f2bf(mean);
    }
    if (t < 31) hbf[(size_t)n * K0PAD + 129 + t] = 0;   // zero pad 129..159
}

// ---------------- pack f32 weights [K,128] into WMMA-B bf16 fragment order --
// fragment element j of lane l maps to K = (j&7) + (j>=8)*16 + (l>>4)*8
__global__ void pack_w128(const float* __restrict__ W, int Kreal, int kchunks,
                          unsigned short* __restrict__ out) {
    int gid = blockIdx.x * blockDim.x + threadIdx.x;
    if (gid >= kchunks * 4096) return;
    int j    = gid & 15;
    int lane = (gid >> 4) & 31;
    int nt   = (gid >> 9) & 7;
    int kc   = gid >> 12;
    int col  = nt * 16 + (lane & 15);
    int kloc = (j & 7) + ((j >> 3) * 16) + ((lane >> 4) * 8);
    int kg   = kc * 32 + kloc;
    float v  = (kg < Kreal) ? W[(size_t)kg * 128 + col] : 0.0f;
    out[gid] = f2bf(v);
}

// ---------------- generic WMMA GEMM (dout=128) + bias + LN + ReLU -----------
// block = 256 threads = 8 waves; wave w -> 16x16 tile at cols [16w,16w+16)
__global__ __launch_bounds__(256)
void gemm128_ln_relu(const unsigned short* __restrict__ A, int lda, int kchunks,
                     const unsigned short* __restrict__ Bp,
                     const float* __restrict__ bias,
                     const float* __restrict__ gam, const float* __restrict__ bet,
                     unsigned short* __restrict__ outb, float* __restrict__ outf) {
    const int t = threadIdx.x, lane = t & 31, w = t >> 5;
    const int row0 = blockIdx.x * 16;
    const int mrow = lane & 15, khalf = lane >> 4;

    v8f c = {};
    for (int kc = 0; kc < kchunks; ++kc) {
        const unsigned short* ap = A + (size_t)(row0 + mrow) * lda + kc * 32 + khalf * 8;
        v8bf alo = *(const v8bf*)ap;
        v8bf ahi = *(const v8bf*)(ap + 16);
        v16bf a = __builtin_shufflevector(alo, ahi,
                    0,1,2,3,4,5,6,7,8,9,10,11,12,13,14,15);
        v16bf b = *(const v16bf*)(Bp + ((size_t)(kc * 8 + w) * 32 + lane) * 16);
        c = __builtin_amdgcn_wmma_f32_16x16x32_bf16(false, a, false, b,
                                                    (short)0, c, false, false);
    }

    __shared__ float tile[16][132];
    __shared__ float mu[16], rs[16];
    const int col = w * 16 + mrow;
    const float bv = bias[col];
#pragma unroll
    for (int i = 0; i < 8; ++i)
        tile[i + khalf * 8][col] = c[i] + bv;      // C layout: VGPR i -> M=i+8*khalf
    __syncthreads();

    if (t < 16) {
        float s = 0.0f;
        for (int k = 0; k < 128; ++k) s += tile[t][k];
        float m = s * (1.0f / 128.0f);
        float v = 0.0f;
        for (int k = 0; k < 128; ++k) { float d = tile[t][k] - m; v += d * d; }
        mu[t] = m;
        rs[t] = rsqrtf(v * (1.0f / 128.0f) + LN_EPS);
    }
    __syncthreads();

    const int r = t >> 4, c0 = (t & 15) * 8;
    const float m = mu[r], q = rs[r];
#pragma unroll
    for (int j = 0; j < 8; ++j) {
        int cc = c0 + j;
        float x = (tile[r][cc] - m) * q * gam[cc] + bet[cc];
        x = fmaxf(x, 0.0f);
        if (outb) outb[(size_t)(row0 + r) * 128 + cc] = f2bf(x);
        if (outf) outf[(size_t)(row0 + r) * 128 + cc] = x;
    }
}

// ---------------- sp_l1 GEMM + LN + ReLU fused with sp_out (128->8) + L2 norm
__global__ __launch_bounds__(256)
void gemm_sp_l1_spout(const unsigned short* __restrict__ A,
                      const unsigned short* __restrict__ Bp,
                      const float* __restrict__ bias,
                      const float* __restrict__ gam, const float* __restrict__ bet,
                      const float* __restrict__ W2, const float* __restrict__ b2,
                      float* __restrict__ sp) {
    const int t = threadIdx.x, lane = t & 31, w = t >> 5;
    const int row0 = blockIdx.x * 16;
    const int mrow = lane & 15, khalf = lane >> 4;

    v8f c = {};
    for (int kc = 0; kc < 4; ++kc) {
        const unsigned short* ap = A + (size_t)(row0 + mrow) * 128 + kc * 32 + khalf * 8;
        v8bf alo = *(const v8bf*)ap;
        v8bf ahi = *(const v8bf*)(ap + 16);
        v16bf a = __builtin_shufflevector(alo, ahi,
                    0,1,2,3,4,5,6,7,8,9,10,11,12,13,14,15);
        v16bf b = *(const v16bf*)(Bp + ((size_t)(kc * 8 + w) * 32 + lane) * 16);
        c = __builtin_amdgcn_wmma_f32_16x16x32_bf16(false, a, false, b,
                                                    (short)0, c, false, false);
    }

    __shared__ float tile[16][132];
    __shared__ float mu[16], rs[16];
    __shared__ float spv[16][8];
    __shared__ float scl[16];
    const int col = w * 16 + mrow;
    const float bv = bias[col];
#pragma unroll
    for (int i = 0; i < 8; ++i)
        tile[i + khalf * 8][col] = c[i] + bv;
    __syncthreads();

    if (t < 16) {
        float s = 0.0f;
        for (int k = 0; k < 128; ++k) s += tile[t][k];
        float m = s * (1.0f / 128.0f);
        float v = 0.0f;
        for (int k = 0; k < 128; ++k) { float d = tile[t][k] - m; v += d * d; }
        mu[t] = m;
        rs[t] = rsqrtf(v * (1.0f / 128.0f) + LN_EPS);
    }
    __syncthreads();

    {   // LN + ReLU in place
        const int r = t >> 4, c0 = (t & 15) * 8;
        const float m = mu[r], q = rs[r];
#pragma unroll
        for (int j = 0; j < 8; ++j) {
            int cc = c0 + j;
            float x = (tile[r][cc] - m) * q * gam[cc] + bet[cc];
            tile[r][cc] = fmaxf(x, 0.0f);
        }
    }
    __syncthreads();

    if (t < 128) {              // sp_out: x2 @ W2[128,8] + b2
        int rr = t >> 3, cc = t & 7;
        float acc = b2[cc];
        for (int k = 0; k < 128; ++k) acc += tile[rr][k] * W2[k * 8 + cc];
        spv[rr][cc] = acc;
    }
    __syncthreads();
    if (t < 16) {               // L2-normalize rows
        float s = 0.0f;
#pragma unroll
        for (int j = 0; j < 8; ++j) s += spv[t][j] * spv[t][j];
        scl[t] = 1.0f / fmaxf(sqrtf(s), 1e-12f);
    }
    __syncthreads();
    if (t < 128) {
        int rr = t >> 3, cc = t & 7;
        sp[(size_t)(row0 + rr) * 8 + cc] = spv[rr][cc] * scl[rr];
    }
}

// ---------------- edge aggregation: one wave32 per edge ---------------------
__global__ __launch_bounds__(256)
void edge_agg(const int* __restrict__ ei, int E,
              const float* __restrict__ sp, const float* __restrict__ h32,
              float* __restrict__ sadd, unsigned* __restrict__ smax,
              float* __restrict__ cnt) {
    int e = (blockIdx.x * blockDim.x + threadIdx.x) >> 5;
    int lane = threadIdx.x & 31;
    if (e >= E) return;
    int s = ei[e], tg = ei[E + e];
    float d = 0.0f;
#pragma unroll
    for (int j = 0; j < 8; ++j) {
        float dd = sp[(size_t)s * 8 + j] - sp[(size_t)tg * 8 + j];
        d += dd * dd;
    }
    float grav = __expf(d * (-1.0f / 0.09f));   // exp(-GRAV_W*d/R^2)
    const float* hs = h32 + (size_t)s * H1P;
    float*    sa = sadd + (size_t)tg * H1P;
    unsigned* sm = smax + (size_t)tg * H1P;
    for (int k = lane; k < H1; k += 32) {
        float v = hs[k] * grav;
        atomicAdd(&sa[k], v);
        unsigned u = __float_as_uint(v);
        u ^= (u & 0x80000000u) ? 0xFFFFFFFFu : 0x80000000u;  // order-preserving
        atomicMax(&sm[k], u);
    }
    if (lane == 0) atomicAdd(&cnt[tg], 1.0f);
}

// ---------------- y = [sum, mean, max, h] in bf16 (pad 516 -> 544) ----------
__global__ __launch_bounds__(256)
void finalize_y(const float* __restrict__ sadd, const unsigned* __restrict__ smax,
                const float* __restrict__ cnt, const float* __restrict__ h32,
                unsigned short* __restrict__ y) {
    int n = blockIdx.x, t = threadIdx.x;
    unsigned short* yr = y + (size_t)n * KYPAD;
    float c = cnt[n];
    if (t < H1) {
        float a = sadd[(size_t)n * H1P + t];
        float mean = a / fmaxf(c, 1.0f);
        float mx = 0.0f;                         // zero-fill for empty segments
        if (c > 0.0f) {
            unsigned u = smax[(size_t)n * H1P + t];
            unsigned bits = (u & 0x80000000u) ? (u ^ 0x80000000u) : ~u;
            mx = __uint_as_float(bits);
        }
        yr[t]           = f2bf(a);
        yr[H1 + t]      = f2bf(mean);
        yr[2 * H1 + t]  = f2bf(mx);
        yr[3 * H1 + t]  = f2bf(h32[(size_t)n * H1P + t]);
    } else if (t < H1 + 28) {
        yr[4 * H1 + (t - H1)] = 0;               // zero pad 516..543
    }
}

// ---------------------------------------------------------------------------
extern "C" void kernel_launch(void* const* d_in, const int* in_sizes, int n_in,
                              void* d_out, int out_size, void* d_ws, size_t ws_size,
                              hipStream_t stream) {
    // --- identify inputs by flat size (robust to flatten-order convention) --
    int idx_hf = 0, idx_ei = -1, idx_ftl0W = -1, idx_spl0W = -1;
    for (int i = 0; i < n_in; ++i) {
        if (in_sizes[i] == N_NODES * H_DIM) idx_hf = i;
        else if (in_sizes[i] == 2 * E_EDGES) idx_ei = i;
        else if (in_sizes[i] == 516 * 128)   idx_ftl0W = i;
        else if (in_sizes[i] == 129 * 128)   idx_spl0W = i;
    }
#define F(i) ((const float*)d_in[(i)])
    const float *spl0W, *spl0b, *spln0g, *spln0b, *spl1W, *spl1b, *spln1g, *spln1b;
    const float *spoutW, *spoutb, *ftl0W, *ftl0b, *ftln0g, *ftln0b;
    const float *ftoutW, *ftoutb, *ftlng, *ftlnb;
    if (idx_spl0W >= 0 && (idx_ftl0W < 0 || idx_spl0W < idx_ftl0W)) {
        // insertion-order flattening: sp_l0{W,b}, sp_ln0{g,b}, sp_l1, sp_ln1,
        // sp_out, ft_l0, ft_ln0, ft_out, ft_lnout
        int q = idx_spl0W;
        spl0W = F(q);      spl0b = F(q + 1);  spln0g = F(q + 2);  spln0b = F(q + 3);
        spl1W = F(q + 4);  spl1b = F(q + 5);  spln1g = F(q + 6);  spln1b = F(q + 7);
        spoutW = F(q + 8); spoutb = F(q + 9);
        ftl0W = F(q + 10); ftl0b = F(q + 11); ftln0g = F(q + 12); ftln0b = F(q + 13);
        ftoutW = F(q + 14); ftoutb = F(q + 15); ftlng = F(q + 16); ftlnb = F(q + 17);
    } else {
        // sorted-pytree flattening: ft_l0{W,b}, ft_ln0{b,g}, ft_lnout{b,g},
        // ft_out{W,b}, sp_l0{W,b}, sp_l1{W,b}, sp_ln0{b,g}, sp_ln1{b,g}, sp_out{W,b}
        int p = (idx_ftl0W >= 0) ? idx_ftl0W : 1;
        ftl0W = F(p);      ftl0b = F(p + 1);
        ftln0b = F(p + 2); ftln0g = F(p + 3);
        ftlnb = F(p + 4);  ftlng = F(p + 5);
        ftoutW = F(p + 6); ftoutb = F(p + 7);
        spl0W = F(p + 8);  spl0b = F(p + 9);
        spl1W = F(p + 10); spl1b = F(p + 11);
        spln0b = F(p + 12); spln0g = F(p + 13);
        spln1b = F(p + 14); spln1g = F(p + 15);
        spoutW = F(p + 16); spoutb = F(p + 17);
    }
    const float* hf = F(idx_hf);
    const int*   ei = (const int*)d_in[(idx_ei >= 0) ? idx_ei : 0];
#undef F

    // --- carve workspace (256B-aligned) ------------------------------------
    char* ws = (char*)d_ws;
    size_t off = 0;
    auto carve = [&](size_t bytes) -> void* {
        off = (off + 255) & ~(size_t)255;
        void* p = ws + off;
        off += bytes;
        return p;
    };
    float*          h32  = (float*)carve((size_t)N_NODES * H1P * 4);
    unsigned short* hbf  = (unsigned short*)carve((size_t)N_NODES * K0PAD * 2);
    unsigned short* x1   = (unsigned short*)carve((size_t)N_NODES * 128 * 2);
    float*          sp   = (float*)carve((size_t)N_NODES * EMB * 4);
    // contiguous accumulator block: sadd | smax | cnt  (all zero-init)
    size_t segf = (size_t)N_NODES * H1P * 2 + N_NODES;
    float*          seg  = (float*)carve(segf * 4);
    float*          sadd = seg;
    unsigned*       smax = (unsigned*)(seg + (size_t)N_NODES * H1P);
    float*          cnt  = seg + (size_t)N_NODES * H1P * 2;
    unsigned short* ybf  = (unsigned short*)carve((size_t)N_NODES * KYPAD * 2);
    unsigned short* zbf  = (unsigned short*)carve((size_t)N_NODES * 128 * 2);
    unsigned short* pw0  = (unsigned short*)carve((size_t)5 * 4096 * 2);
    unsigned short* pw1  = (unsigned short*)carve((size_t)4 * 4096 * 2);
    unsigned short* pwf0 = (unsigned short*)carve((size_t)17 * 4096 * 2);
    unsigned short* pwf1 = (unsigned short*)carve((size_t)4 * 4096 * 2);
    (void)ws_size; (void)out_size; (void)n_in;

    // --- pipeline -----------------------------------------------------------
    zero_f32<<<2048, 256, 0, stream>>>(seg, (long long)segf);
    prep_h<<<N_NODES, 128, 0, stream>>>(hf, h32, hbf);

    pack_w128<<<(5 * 4096 + 255) / 256, 256, 0, stream>>>(spl0W, 129, 5, pw0);
    pack_w128<<<(4 * 4096 + 255) / 256, 256, 0, stream>>>(spl1W, 128, 4, pw1);
    pack_w128<<<(17 * 4096 + 255) / 256, 256, 0, stream>>>(ftl0W, 516, 17, pwf0);
    pack_w128<<<(4 * 4096 + 255) / 256, 256, 0, stream>>>(ftoutW, 128, 4, pwf1);

    // spatial MLP
    gemm128_ln_relu<<<N_NODES / 16, 256, 0, stream>>>(
        hbf, K0PAD, 5, pw0, spl0b, spln0g, spln0b, x1, nullptr);
    gemm_sp_l1_spout<<<N_NODES / 16, 256, 0, stream>>>(
        x1, pw1, spl1b, spln1g, spln1b, spoutW, spoutb, sp);

    // edge gravity scatter (sum / max / count) into L2-resident accumulators
    edge_agg<<<E_EDGES / 8, 256, 0, stream>>>(ei, E_EDGES, sp, h32, sadd, smax, cnt);
    finalize_y<<<N_NODES, 256, 0, stream>>>(sadd, smax, cnt, h32, ybf);

    // feature MLP
    gemm128_ln_relu<<<N_NODES / 16, 256, 0, stream>>>(
        ybf, KYPAD, 17, pwf0, ftl0b, ftln0g, ftln0b, zbf, nullptr);
    gemm128_ln_relu<<<N_NODES / 16, 256, 0, stream>>>(
        zbf, 128, 4, pwf1, ftoutb, ftlng, ftlnb, nullptr, (float*)d_out);
}
